// SelfAttention_7438883357604
// MI455X (gfx1250) — matrix operand order repaired and checked by reading, not compiled, not run on hardware
//
#include <hip/hip_runtime.h>

typedef __attribute__((ext_vector_type(16))) __bf16       v16bf;
typedef __attribute__((ext_vector_type(8)))  float        v8f;
typedef __attribute__((ext_vector_type(4)))  unsigned int u32x4;

union Frag16 { u32x4 u[2]; v16bf v; };

#define B_     2
#define S_     2048
#define HID_   256
#define HEADS_ 8
#define HD_    32

// ---------------------------------------------------------------- convert
__global__ void cvt_f32_bf16(const float* __restrict__ src, __bf16* __restrict__ dst, int n) {
  int i = blockIdx.x * blockDim.x + threadIdx.x;
  if (i < n) dst[i] = (__bf16)__builtin_nontemporal_load(src + i);  // read-once stream
}

// ---------------------------------------------------------------- GEMM  C[M,N] = X[M,K] * W[N,K]^T + bias
// mode 0: store bf16 to [b,h,s,d]   (Q / K), scaled
// mode 1: store bf16 to [b,h,d,s]   (V transposed)
// mode 2: store f32  row-major      (final projection, NT store)
__global__ void gemm_bf16_wmma(const __bf16* __restrict__ X, const __bf16* __restrict__ W,
                               const float* __restrict__ bias, __bf16* __restrict__ obf,
                               float* __restrict__ of32, int mode, float scale)
{
  const int lane = threadIdx.x & 31;
  const int wave = threadIdx.x >> 5;
  const int l16  = lane & 15;
  const int half = lane >> 4;
  const int tile = blockIdx.x * (blockDim.x >> 5) + wave;   // 4096 tiles
  const int mt = tile >> 4;                                 // 256 M-tiles
  const int nt = tile & 15;                                 // 16  N-tiles

  const __bf16* arow = X + (size_t)(mt * 16 + l16) * HID_;
  const __bf16* brow = W + (size_t)(nt * 16 + l16) * HID_;

  v8f acc = {};
#pragma unroll
  for (int ks = 0; ks < HID_; ks += 32) {
    Frag16 A, Bm;
    // 16-bit A layout: lane m=l16; halfs [half*8 .. +7] and [16+half*8 .. +7]
    A.u[0]  = *reinterpret_cast<const u32x4*>(arow + ks + half * 8);
    A.u[1]  = *reinterpret_cast<const u32x4*>(arow + ks + 16 + half * 8);
    // 16-bit B layout: lane n=l16; K halfs [half*8 .. +7] and [16+half*8 .. +7] (B[k][n] = W[n][k])
    Bm.u[0] = *reinterpret_cast<const u32x4*>(brow + ks + half * 8);
    Bm.u[1] = *reinterpret_cast<const u32x4*>(brow + ks + half * 8 + 16);
    acc = __builtin_amdgcn_wmma_f32_16x16x32_bf16(false, A.v, false, Bm.v,
                                                  (short)0, acc, false, false);
  }

  const int col = nt * 16 + l16;
  const float bv = bias[col];
#pragma unroll
  for (int r = 0; r < 8; ++r) {
    const int m = mt * 16 + r + 8 * half;      // C/D layout: lanes0-15->M=r, lanes16-31->M=r+8
    const float val = (acc[r] + bv) * scale;
    if (mode == 2) {
      __builtin_nontemporal_store(val, of32 + (size_t)m * HID_ + col);  // final output: write-once
    } else {
      const int b = m >> 11, s = m & (S_ - 1);
      const int h = col >> 5, d = col & 31;
      size_t idx;
      if (mode == 0) idx = (((size_t)(b * HEADS_ + h)) * S_ + s) * HD_ + d;  // [b,h,s,d]
      else           idx = (((size_t)(b * HEADS_ + h)) * HD_ + d) * S_ + s;  // [b,h,d,s]
      obf[idx] = (__bf16)val;
    }
  }
}

// ---------------------------------------------------------------- flash attention
__global__ void attn_bf16_wmma(const __bf16* __restrict__ Q, const __bf16* __restrict__ K,
                               const __bf16* __restrict__ Vt, const float* __restrict__ pos,
                               const int* __restrict__ mask, __bf16* __restrict__ AO)
{
  __shared__ __attribute__((aligned(16))) __bf16 plds[8][16 * 40];  // padded rows, 80B stride

  const int lane = threadIdx.x & 31;
  const int wave = threadIdx.x >> 5;
  const int l16  = lane & 15;
  const int half = lane >> 4;
  const int tile = blockIdx.x * (blockDim.x >> 5) + wave;   // 2048 tiles
  const int qt = tile & 127;
  const int bh = tile >> 7;            // b*8+h
  const int b  = bh >> 3;
  const int q0 = qt * 16;

  // Q A-fragment (16 x 32 bf16 == HD) loaded once, 1/sqrt(HD) folded in upstream
  Frag16 Qa;
  {
    const __bf16* qrow = Q + ((size_t)bh * S_ + q0 + l16) * HD_;
    Qa.u[0] = *reinterpret_cast<const u32x4*>(qrow + half * 8);
    Qa.u[1] = *reinterpret_cast<const u32x4*>(qrow + 16 + half * 8);
  }

  v8f O0 = {}, O1 = {};
  float mx[8], sm[8];
#pragma unroll
  for (int r = 0; r < 8; ++r) { mx[r] = -1e30f; sm[r] = 0.f; }

  const float* posq = pos + ((size_t)bh * S_ + q0) * S_;
  const int*   mrow = mask + b * S_;
  __bf16*      pw   = &plds[wave][0];

  for (int kb = 0; kb < S_; kb += 32) {
    // K^T B-fragments: B[d][n] = K[b,h, kb+n, d], rows contiguous in d
    Frag16 Bk0, Bk1;
    const __bf16* krow0 = K + ((size_t)bh * S_ + kb + l16) * HD_ + half * 8;
    const __bf16* krow1 = krow0 + 16 * HD_;
    Bk0.u[0] = *reinterpret_cast<const u32x4*>(krow0);
    Bk0.u[1] = *reinterpret_cast<const u32x4*>(krow0 + 16);
    Bk1.u[0] = *reinterpret_cast<const u32x4*>(krow1);
    Bk1.u[1] = *reinterpret_cast<const u32x4*>(krow1 + 16);

    const v8f z = {};
    v8f E0 = __builtin_amdgcn_wmma_f32_16x16x32_bf16(false, Qa.v, false, Bk0.v, (short)0, z, false, false);
    v8f E1 = __builtin_amdgcn_wmma_f32_16x16x32_bf16(false, Qa.v, false, Bk1.v, (short)0, z, false, false);

    const int mk0 = mrow[kb + l16];
    const int mk1 = mrow[kb + 16 + l16];
    if (kb + 32 < S_) __builtin_prefetch(posq + kb + 32, 0, 0);  // global_prefetch_b8

#pragma unroll
    for (int r = 0; r < 8; ++r) {
      const int qm = r + 8 * half;   // this lane's row in the C/D tile
      // pos is a 268MB read-once stream: NT hint keeps it from sweeping L2,
      // preserving the L2-resident K/V tiles that all 128 q-tiles re-read.
      float e0 = E0[r] + __builtin_nontemporal_load(posq + (size_t)qm * S_ + kb + l16);
      float e1 = E1[r] + __builtin_nontemporal_load(posq + (size_t)qm * S_ + kb + 16 + l16);
      if (mk0 == 0) e0 = -1e30f;
      if (mk1 == 0) e1 = -1e30f;

      float rmx = fmaxf(e0, e1);
#pragma unroll
      for (int xm = 1; xm < 16; xm <<= 1) rmx = fmaxf(rmx, __shfl_xor(rmx, xm, 32));
      const float nm   = fmaxf(mx[r], rmx);
      const float corr = __expf(mx[r] - nm);
      mx[r] = nm;
      const float p0 = __expf(e0 - nm);
      const float p1 = __expf(e1 - nm);
      float rs = p0 + p1;
#pragma unroll
      for (int xm = 1; xm < 16; xm <<= 1) rs += __shfl_xor(rs, xm, 32);
      sm[r] = sm[r] * corr + rs;
      O0[r] *= corr;
      O1[r] *= corr;
      // stage P (C/D layout) into LDS so it can be re-read in A layout
      pw[qm * 40 + l16]      = (__bf16)p0;
      pw[qm * 40 + 16 + l16] = (__bf16)p1;
    }

    asm volatile("s_wait_dscnt 0" ::: "memory");   // DS store -> DS load ordering

    Frag16 Pa;   // P as 16x32 A-fragment
    Pa.u[0] = *reinterpret_cast<const u32x4*>(pw + l16 * 40 + half * 8);
    Pa.u[1] = *reinterpret_cast<const u32x4*>(pw + l16 * 40 + 16 + half * 8);

    // V B-fragments from transposed V: B[k][n] = Vt[b,h, n(+16), kb+k]
    Frag16 Bv0, Bv1;
    const __bf16* vrow0 = Vt + ((size_t)bh * HD_ + l16) * S_ + kb + half * 8;
    const __bf16* vrow1 = vrow0 + 16 * S_;
    Bv0.u[0] = *reinterpret_cast<const u32x4*>(vrow0);
    Bv0.u[1] = *reinterpret_cast<const u32x4*>(vrow0 + 16);
    Bv1.u[0] = *reinterpret_cast<const u32x4*>(vrow1);
    Bv1.u[1] = *reinterpret_cast<const u32x4*>(vrow1 + 16);

    O0 = __builtin_amdgcn_wmma_f32_16x16x32_bf16(false, Pa.v, false, Bv0.v, (short)0, O0, false, false);
    O1 = __builtin_amdgcn_wmma_f32_16x16x32_bf16(false, Pa.v, false, Bv1.v, (short)0, O1, false, false);
  }

  // normalize and store to [b, s, h, d] (== [B,S,HID] after head concat)
#pragma unroll
  for (int r = 0; r < 8; ++r) {
    const int qm = r + 8 * half;
    const int s  = q0 + qm;
    const float inv = 1.0f / sm[r];
    const size_t base = (((size_t)b * S_ + s) * HEADS_ + (bh & 7)) * HD_;
    AO[base + l16]      = (__bf16)(O0[r] * inv);
    AO[base + 16 + l16] = (__bf16)(O1[r] * inv);
  }
}

// ---------------------------------------------------------------- launch
extern "C" void kernel_launch(void* const* d_in, const int* in_sizes, int n_in,
                              void* d_out, int out_size, void* d_ws, size_t ws_size,
                              hipStream_t stream)
{
  (void)in_sizes; (void)n_in; (void)out_size; (void)ws_size;
  const float* q    = (const float*)d_in[0];
  const float* k    = (const float*)d_in[1];
  const float* v    = (const float*)d_in[2];
  const float* pos  = (const float*)d_in[3];
  const int*   msk  = (const int*)  d_in[4];
  const float* Wq   = (const float*)d_in[5];
  const float* bq   = (const float*)d_in[6];
  const float* Wk   = (const float*)d_in[7];
  const float* bk   = (const float*)d_in[8];
  const float* Wv   = (const float*)d_in[9];
  const float* bv   = (const float*)d_in[10];
  const float* Wfc  = (const float*)d_in[11];
  const float* bfc  = (const float*)d_in[12];

  const size_t NX = (size_t)B_ * S_ * HID_;   // 1,048,576
  const size_t NW = (size_t)HID_ * HID_;      // 65,536

  __bf16* w   = (__bf16*)d_ws;
  __bf16 *Xq  = w,        *Xk  = Xq  + NX, *Xv  = Xk  + NX;
  __bf16 *Wqb = Xv + NX,  *Wkb = Wqb + NW, *Wvb = Wkb + NW, *Wfb = Wvb + NW;
  __bf16 *Qb  = Wfb + NW, *Kb  = Qb  + NX, *Vtb = Kb  + NX, *AO  = Vtb + NX;

  const int T = 256;
  cvt_f32_bf16<<<(int)(NX / T), T, 0, stream>>>(q,   Xq,  (int)NX);
  cvt_f32_bf16<<<(int)(NX / T), T, 0, stream>>>(k,   Xk,  (int)NX);
  cvt_f32_bf16<<<(int)(NX / T), T, 0, stream>>>(v,   Xv,  (int)NX);
  cvt_f32_bf16<<<(int)(NW / T), T, 0, stream>>>(Wq,  Wqb, (int)NW);
  cvt_f32_bf16<<<(int)(NW / T), T, 0, stream>>>(Wk,  Wkb, (int)NW);
  cvt_f32_bf16<<<(int)(NW / T), T, 0, stream>>>(Wv,  Wvb, (int)NW);
  cvt_f32_bf16<<<(int)(NW / T), T, 0, stream>>>(Wfc, Wfb, (int)NW);

  const float invscale = 0.17677669529663687f;  // 1/sqrt(HD)=1/sqrt(32)
  // 4096 tiles / 8 waves per block = 512 blocks
  gemm_bf16_wmma<<<512, 256, 0, stream>>>(Xq, Wqb, bq, Qb,  nullptr, 0, invscale);
  gemm_bf16_wmma<<<512, 256, 0, stream>>>(Xk, Wkb, bk, Kb,  nullptr, 0, 1.0f);
  gemm_bf16_wmma<<<512, 256, 0, stream>>>(Xv, Wvb, bv, Vtb, nullptr, 1, 1.0f);

  // 2048 attention tiles / 8 waves = 256 blocks
  attn_bf16_wmma<<<256, 256, 0, stream>>>(Qb, Kb, Vtb, pos, msk, AO);

  gemm_bf16_wmma<<<512, 256, 0, stream>>>(AO, Wfb, bfc, nullptr, (float*)d_out, 2, 1.0f);
}
